// LinBnA_int_31628139168645
// MI455X (gfx1250) — 8-bit matrix operand order repaired and checked by reading, compiled, not run on hardware
//
#include <hip/hip_runtime.h>
#include <stdint.h>

typedef __attribute__((ext_vector_type(8))) int v8i;
typedef __attribute__((ext_vector_type(4))) int v4i;
typedef __attribute__((ext_vector_type(2))) int v2i;

#define M_TOT 8192
#define K_TOT 4096
#define N_TOT 4096
#define TK   128
#define LDSK (TK + 16)

#define TMA 256
#define TNA 128
#define A_BYTES (TMA * LDSK)
#define B_BYTES (TNA * LDSK)
#define PBUF (A_BYTES + B_BYTES)
#define NS (K_TOT / TK)

#define TMF 128
#define TNF 128
#define FA_BYTES (TMF * LDSK)
#define FBUF (2 * FA_BYTES)

#if defined(__has_builtin)
#  if __has_builtin(__builtin_amdgcn_s_wait_asynccnt)
#    define WAIT_ASYNC(n) __builtin_amdgcn_s_wait_asynccnt(n)
#  endif
#endif
#ifndef WAIT_ASYNC
#  define WAIT_ASYNC(n) asm volatile("s_wait_asynccnt %0" ::"i"(n) : "memory")
#endif

__device__ __forceinline__ void async16(uint32_t lds_off, const signed char* g) {
  asm volatile("global_load_async_to_lds_b128 %0, %1, off" ::"v"(lds_off), "v"(g)
               : "memory");
}

__device__ __forceinline__ v4i pack16(const int* __restrict__ g) {
  v4i a = *(const v4i*)(g);
  v4i b = *(const v4i*)(g + 4);
  v4i c = *(const v4i*)(g + 8);
  v4i d = *(const v4i*)(g + 12);
  v4i r;
  r[0] = (a[0] & 255) | ((a[1] & 255) << 8) | ((a[2] & 255) << 16) | (a[3] << 24);
  r[1] = (b[0] & 255) | ((b[1] & 255) << 8) | ((b[2] & 255) << 16) | (b[3] << 24);
  r[2] = (c[0] & 255) | ((c[1] & 255) << 8) | ((c[2] & 255) << 16) | (c[3] << 24);
  r[3] = (d[0] & 255) | ((d[1] & 255) << 8) | ((d[2] & 255) << 16) | (d[3] << 24);
  return r;
}

__global__ __launch_bounds__(256) void pack_i32_to_i8(const int* __restrict__ in,
                                                      signed char* __restrict__ outp,
                                                      int n16) {
  int i = blockIdx.x * 256 + threadIdx.x;
  if (i < n16) *(v4i*)(outp + (size_t)i * 16) = pack16(in + (size_t)i * 16);
}

__global__ __launch_bounds__(256) void lin_bn_i8_wmma_async(
    const signed char* __restrict__ xp, const signed char* __restrict__ wp,
    const int* __restrict__ tvec, const int* __restrict__ nvec,
    const int* __restrict__ aminv, const int* __restrict__ amaxv,
    float* __restrict__ out)
{
  __shared__ unsigned char smem[2 * PBUF];
  const int tid  = threadIdx.x;
  const int lane = tid & 31;
  const int wv   = tid >> 5;
  const int half = lane >> 4;
  const int l    = lane & 15;
  const int rmBase = (wv >> 1) * 64;
  const int cnBase = (wv & 1) * 64;
  const int bm = blockIdx.x;
  const int bn = blockIdx.y;

  const int rA = tid >> 3;
  const int cA = (tid & 7) * 16;

  const signed char* xg = xp + (size_t)(bm * TMA + rA) * K_TOT + cA;
  const signed char* wg = wp + (size_t)(bn * TNA + rA) * K_TOT + cA;
  const uint32_t la0 = (uint32_t)(uintptr_t)(&smem[0]) + rA * LDSK + cA;
  const uint32_t lb0 = la0 + A_BYTES;

  v8i acc[4][4];
  #pragma unroll
  for (int m = 0; m < 4; ++m)
    #pragma unroll
    for (int j = 0; j < 4; ++j)
      #pragma unroll
      for (int e = 0; e < 8; ++e) acc[m][j][e] = 0;

  #pragma unroll
  for (int r = 0; r < 8; ++r)
    async16(la0 + r * 32 * LDSK, xg + (size_t)(r * 32) * K_TOT);
  #pragma unroll
  for (int r = 0; r < 4; ++r)
    async16(lb0 + r * 32 * LDSK, wg + (size_t)(r * 32) * K_TOT);

  #pragma unroll 1
  for (int s = 0; s < NS; ++s) {
    const int buf = s & 1;
    if (s + 1 < NS) {
      const int k0 = (s + 1) * TK;
      const uint32_t la = la0 + (buf ^ 1) * PBUF;
      const uint32_t lb = lb0 + (buf ^ 1) * PBUF;
      #pragma unroll
      for (int r = 0; r < 8; ++r)
        async16(la + r * 32 * LDSK, xg + k0 + (size_t)(r * 32) * K_TOT);
      #pragma unroll
      for (int r = 0; r < 4; ++r)
        async16(lb + r * 32 * LDSK, wg + k0 + (size_t)(r * 32) * K_TOT);
      WAIT_ASYNC(12);
    } else {
      WAIT_ASYNC(0);
    }
    __syncthreads();

    const unsigned char* sA = smem + buf * PBUF;
    const unsigned char* sB = sA + A_BYTES;
    #pragma unroll
    for (int kk = 0; kk < TK; kk += 64) {
      v8i af[4], bf[4];
      #pragma unroll
      for (int m = 0; m < 4; ++m) {
        const unsigned char* p = sA + (rmBase + m * 16 + l) * LDSK + kk + half * 8;
        v2i q0 = *(const v2i*)(p);
        v2i q1 = *(const v2i*)(p + 16);
        v2i q2 = *(const v2i*)(p + 32);
        v2i q3 = *(const v2i*)(p + 48);
        af[m][0] = q0[0]; af[m][1] = q0[1];
        af[m][2] = q1[0]; af[m][3] = q1[1];
        af[m][4] = q2[0]; af[m][5] = q2[1];
        af[m][6] = q3[0]; af[m][7] = q3[1];
      }
      #pragma unroll
      for (int j = 0; j < 4; ++j) {
        const unsigned char* p = sB + (cnBase + j * 16 + l) * LDSK + kk + half * 8;
        v2i r0 = *(const v2i*)(p);
        v2i r1 = *(const v2i*)(p + 16);
        v2i r2 = *(const v2i*)(p + 32);
        v2i r3 = *(const v2i*)(p + 48);
        bf[j][0] = r0[0]; bf[j][1] = r0[1]; bf[j][2] = r1[0]; bf[j][3] = r1[1];
        bf[j][4] = r2[0]; bf[j][5] = r2[1]; bf[j][6] = r3[0]; bf[j][7] = r3[1];
      }
      #pragma unroll
      for (int m = 0; m < 4; ++m)
        #pragma unroll
        for (int j = 0; j < 4; ++j)
          acc[m][j] = __builtin_amdgcn_wmma_i32_16x16x64_iu8(
              true, af[m], true, bf[j], acc[m][j], false, false);
    }
    __syncthreads();
  }

  #pragma unroll
  for (int j = 0; j < 4; ++j) {
    const int col = bn * TNA + cnBase + j * 16 + l;
    const int tb = tvec[col];
    const int sh = -nvec[col];
    const int lo = aminv[col];
    const int hi = amaxv[col];
    #pragma unroll
    for (int m = 0; m < 4; ++m) {
      const int row0 = bm * TMA + rmBase + m * 16 + half * 8;
      #pragma unroll
      for (int e = 0; e < 8; ++e) {
        int v = acc[m][j][e] + tb;
        v = v >> sh;
        v = v < lo ? lo : v;
        v = v > hi ? hi : v;
        out[(size_t)(row0 + e) * N_TOT + col] = (float)v;
      }
    }
  }
}

__global__ __launch_bounds__(256) void lin_bn_i8_wmma_fused(
    const int* __restrict__ x, const int* __restrict__ w,
    const int* __restrict__ tvec, const int* __restrict__ nvec,
    const int* __restrict__ aminv, const int* __restrict__ amaxv,
    float* __restrict__ out)
{
  __shared__ unsigned char smem[FBUF];
  const int tid  = threadIdx.x;
  const int lane = tid & 31;
  const int wv   = tid >> 5;
  const int half = lane >> 4;
  const int l    = lane & 15;
  const int rmBase = (wv >> 2) * 64;
  const int cnBase = (wv & 3) * 32;
  const int bm = blockIdx.x;
  const int bn = blockIdx.y;
  const int rA = tid >> 3;
  const int cA = (tid & 7) * 16;

  const int* xg = x + (size_t)(bm * TMF + rA) * K_TOT + cA;
  const int* wg = w + (size_t)(bn * TNF + rA) * K_TOT + cA;
  unsigned char* sAw = smem + rA * LDSK + cA;
  unsigned char* sBw = smem + FA_BYTES + rA * LDSK + cA;

  v8i acc[4][2];
  #pragma unroll
  for (int m = 0; m < 4; ++m)
    #pragma unroll
    for (int j = 0; j < 2; ++j)
      #pragma unroll
      for (int e = 0; e < 8; ++e) acc[m][j][e] = 0;

  v4i stA[4], stB[4];
  #pragma unroll
  for (int r = 0; r < 4; ++r) {
    stA[r] = pack16(xg + (size_t)(r * 32) * K_TOT);
    stB[r] = pack16(wg + (size_t)(r * 32) * K_TOT);
  }

  #pragma unroll 1
  for (int s = 0; s < NS; ++s) {
    __syncthreads();
    #pragma unroll
    for (int r = 0; r < 4; ++r) {
      *(v4i*)(sAw + r * 32 * LDSK) = stA[r];
      *(v4i*)(sBw + r * 32 * LDSK) = stB[r];
    }
    if (s + 1 < NS) {
      const int k0 = (s + 1) * TK;
      #pragma unroll
      for (int r = 0; r < 4; ++r) {
        stA[r] = pack16(xg + k0 + (size_t)(r * 32) * K_TOT);
        stB[r] = pack16(wg + k0 + (size_t)(r * 32) * K_TOT);
      }
    }
    __syncthreads();

    #pragma unroll
    for (int kk = 0; kk < TK; kk += 64) {
      v8i af[4], bf[2];
      #pragma unroll
      for (int m = 0; m < 4; ++m) {
        const unsigned char* p = smem + (rmBase + m * 16 + l) * LDSK + kk + half * 8;
        v2i q0 = *(const v2i*)(p);
        v2i q1 = *(const v2i*)(p + 16);
        v2i q2 = *(const v2i*)(p + 32);
        v2i q3 = *(const v2i*)(p + 48);
        af[m][0] = q0[0]; af[m][1] = q0[1];
        af[m][2] = q1[0]; af[m][3] = q1[1];
        af[m][4] = q2[0]; af[m][5] = q2[1];
        af[m][6] = q3[0]; af[m][7] = q3[1];
      }
      #pragma unroll
      for (int j = 0; j < 2; ++j) {
        const unsigned char* p = smem + FA_BYTES + (cnBase + j * 16 + l) * LDSK + kk + half * 8;
        v2i r0 = *(const v2i*)(p);
        v2i r1 = *(const v2i*)(p + 16);
        v2i r2 = *(const v2i*)(p + 32);
        v2i r3 = *(const v2i*)(p + 48);
        bf[j][0] = r0[0]; bf[j][1] = r0[1]; bf[j][2] = r1[0]; bf[j][3] = r1[1];
        bf[j][4] = r2[0]; bf[j][5] = r2[1]; bf[j][6] = r3[0]; bf[j][7] = r3[1];
      }
      #pragma unroll
      for (int m = 0; m < 4; ++m)
        #pragma unroll
        for (int j = 0; j < 2; ++j)
          acc[m][j] = __builtin_amdgcn_wmma_i32_16x16x64_iu8(
              true, af[m], true, bf[j], acc[m][j], false, false);
    }
  }

  #pragma unroll
  for (int j = 0; j < 2; ++j) {
    const int col = bn * TNF + cnBase + j * 16 + l;
    const int tb = tvec[col];
    const int sh = -nvec[col];
    const int lo = aminv[col];
    const int hi = amaxv[col];
    #pragma unroll
    for (int m = 0; m < 4; ++m) {
      const int row0 = bm * TMF + rmBase + m * 16 + half * 8;
      #pragma unroll
      for (int e = 0; e < 8; ++e) {
        int v = acc[m][j][e] + tb;
        v = v >> sh;
        v = v < lo ? lo : v;
        v = v > hi ? hi : v;
        out[(size_t)(row0 + e) * N_TOT + col] = (float)v;
      }
    }
  }
}

extern "C" void kernel_launch(void* const* d_in, const int* in_sizes, int n_in,
                              void* d_out, int out_size, void* d_ws, size_t ws_size,
                              hipStream_t stream) {
  const int* x    = (const int*)d_in[0];
  const int* w    = (const int*)d_in[1];
  const int* t    = (const int*)d_in[2];
  const int* n    = (const int*)d_in[3];
  const int* amin = (const int*)d_in[4];
  const int* amax = (const int*)d_in[5];
  float* out = (float*)d_out;

  const size_t XB = (size_t)M_TOT * K_TOT;
  const size_t WB = (size_t)N_TOT * K_TOT;
  if (ws_size >= XB + WB) {
    signed char* xp = (signed char*)d_ws;
    signed char* wp = xp + XB;
    const int xn16 = (int)(XB / 16);
    const int wn16 = (int)(WB / 16);
    pack_i32_to_i8<<<(xn16 + 255) / 256, 256, 0, stream>>>(x, xp, xn16);
    pack_i32_to_i8<<<(wn16 + 255) / 256, 256, 0, stream>>>(w, wp, wn16);
    dim3 grid(M_TOT / TMA, N_TOT / TNA);
    lin_bn_i8_wmma_async<<<grid, dim3(256), 0, stream>>>(xp, wp, t, n, amin, amax, out);
  } else {
    dim3 grid(M_TOT / TMF, N_TOT / TNF);
    lin_bn_i8_wmma_fused<<<grid, dim3(256), 0, stream>>>(x, w, t, n, amin, amax, out);
  }
  (void)in_sizes; (void)n_in; (void)out_size;
}
